// SHTemplateRefinerHard_84464826843895
// MI455X (gfx1250) — matrix operand order repaired and checked by reading, not compiled, not run on hardware
//
#include <hip/hip_runtime.h>
#include <hip/hip_bf16.h>
#include <stdint.h>

#define GMAXC 4
#define EMAXC 10
#define NATOMC 14
#define AAC 20
#define HD 256
#define KP1 288   // padded K for layer 1: 256 (h) + 20 (probs) + 12 zero
#define K2 256

typedef __attribute__((ext_vector_type(16))) __bf16        v16bf;
typedef __attribute__((ext_vector_type(8)))  float         v8f;
typedef __attribute__((ext_vector_type(4)))  unsigned int  uint4v;
typedef __attribute__((ext_vector_type(4)))  float         float4v;

union Frag {
  v16bf  v;
  uint4v q[2];
};

static __device__ __forceinline__ v8f wmma_bf16(const v16bf& a, const v16bf& b, const v8f& c) {
  // D = A(16x32 bf16) x B(32x16 bf16) + C(16x16 f32)
  return __builtin_amdgcn_wmma_f32_16x16x32_bf16(false, a, false, b, (short)0, c, false, false);
}

static __device__ __forceinline__ unsigned short f2bf_bits(float x) {
  __bf16 b = (__bf16)x;
  return __builtin_bit_cast(unsigned short, b);
}

// Async global->LDS 16-byte copy (CDNA5, tracked by ASYNCcnt).
// lds_off = workgroup-relative LDS byte offset; gptr = global address.
static __device__ __forceinline__ void async_load16(unsigned lds_off, const void* gptr) {
  asm volatile("global_load_async_to_lds_b128 %0, %1, off"
               :: "v"(lds_off), "v"((unsigned long long)(uintptr_t)gptr)
               : "memory");
}
static __device__ __forceinline__ void wait_async0() {
  asm volatile("s_wait_asynccnt 0x0" ::: "memory");
}

// ---------------------------------------------------------------------------
// One-shot weight pack: bf16, column-major [n][k] so a lane's B-fragment is two 16B
// runs along k, 32B apart.  W1T: 256x288 (k>=276 zero).  W2T: 256x256.
// WHT: 32x256 (n<4 -> Wt col n ; 16<=n<30 -> Ws col n-16 ; else 0).
// ---------------------------------------------------------------------------
__global__ void pack_weights_kernel(const float* __restrict__ W1, const float* __restrict__ W2,
                                    const float* __restrict__ Wt, const float* __restrict__ Ws,
                                    unsigned short* __restrict__ w1t,
                                    unsigned short* __restrict__ w2t,
                                    unsigned short* __restrict__ wht) {
  const int N1 = HD * KP1;   // 73728
  const int N2 = HD * K2;    // 65536
  const int N3 = 32 * K2;    // 8192
  int idx = blockIdx.x * blockDim.x + threadIdx.x;
  if (idx < N1) {
    int n = idx / KP1, k = idx % KP1;
    float v = (k < HD + AAC) ? W1[k * HD + n] : 0.f;
    w1t[idx] = f2bf_bits(v);
  } else if (idx < N1 + N2) {
    int j = idx - N1;
    int n = j / K2, k = j % K2;
    w2t[j] = f2bf_bits(W2[k * HD + n]);
  } else if (idx < N1 + N2 + N3) {
    int j = idx - N1 - N2;
    int n = j / K2, k = j % K2;
    float v = 0.f;
    if (n < GMAXC)                       v = Wt[k * GMAXC + n];
    else if (n >= 16 && n < 16 + NATOMC) v = Ws[k * NATOMC + (n - 16)];
    wht[j] = f2bf_bits(v);
  }
}

// ---------------------------------------------------------------------------
// Fused MLP: 8 waves/block, 128 rows/block; each wave = 16 rows x 256 cols.
// Weight K-slices are double-buffer staged into LDS with async copies so the
// next 16KB slice streams in while WMMAs consume the current one.
// ---------------------------------------------------------------------------
__global__ void __launch_bounds__(256, 1) mlp_heads_kernel(
    const float* __restrict__ hin, const float* __restrict__ probs,
    const float* __restrict__ b1, const float* __restrict__ b2,
    const float* __restrict__ bt, const float* __restrict__ bs,
    const unsigned short* __restrict__ w1t_u, const unsigned short* __restrict__ w2t_u,
    const unsigned short* __restrict__ wht_u,
    float* __restrict__ dchi, float* __restrict__ sraw) {
  __shared__ unsigned short zbuf_u[128 * 256];    // 64 KB activation tile (bf16)
  __shared__ unsigned short wbuf_u[2][256 * 32];  // 2 x 16 KB weight K-slices
  __bf16* zb = reinterpret_cast<__bf16*>(zbuf_u);
  const __bf16* W1T = reinterpret_cast<const __bf16*>(w1t_u);
  const __bf16* W2T = reinterpret_cast<const __bf16*>(w2t_u);
  const __bf16* WHT = reinterpret_cast<const __bf16*>(wht_u);

  const int  lane = threadIdx.x & 31;
  const int  wave = threadIdx.x >> 5;
  const int  col  = lane & 15;          // N (or M for A-frag) within tile
  const bool hi   = lane >= 16;
  const int  rowBase = blockIdx.x * 128 + wave * 16;

  const unsigned wb_base[2] = {(unsigned)(uintptr_t)&wbuf_u[0][0],
                               (unsigned)(uintptr_t)&wbuf_u[1][0]};

  // Stage one 32-K weight slice (256 cols x 32 k bf16 = 16KB) into LDS buffer.
  // Each of the 256 threads copies one 64B row with 4 async b128 transfers.
  auto stage_slice = [&](const __bf16* Wsrc, int ldK, int kb, int buf) {
    const int n = threadIdx.x;
    const __bf16* src = Wsrc + (size_t)n * ldK + kb * 32;
    const unsigned dst = wb_base[buf] + (unsigned)(n * 32) * 2u;
#pragma unroll
    for (int j = 0; j < 4; ++j) async_load16(dst + j * 16u, src + j * 8);
  };

  const v8f vzero = {0.f, 0.f, 0.f, 0.f, 0.f, 0.f, 0.f, 0.f};
  v8f acc[16];

  // ----------------- Phase 1: z1 = silu([h|probs] @ W1 + b1) -----------------
#pragma unroll
  for (int t = 0; t < 16; ++t) acc[t] = vzero;

  stage_slice(W1T, KP1, 0, 0);
  wait_async0();
  __syncthreads();
  {
    const int    arow = rowBase + col;                 // A-frag row (M = lane&15)
    const float* hrow = hin   + (size_t)arow * HD;
    const float* prow = probs + (size_t)arow * AAC;
    for (int kb = 0; kb < KP1 / 32; ++kb) {
      const int buf = kb & 1;
      if (kb + 1 < KP1 / 32) stage_slice(W1T, KP1, kb + 1, 1 - buf);

      Frag a;
      const int k0 = kb * 32 + (hi ? 8 : 0);           // A layout: two 8-K chunks
      if (kb < 8) {                                    // all from h, vectorized
        const float4v* p0 = reinterpret_cast<const float4v*>(hrow + k0);
        float4v c0a = p0[0], c0b = p0[1];              // K = k0 .. k0+7
        float4v c1a = p0[4], c1b = p0[5];              // K = k0+16 .. k0+23
        a.v[0] = (__bf16)c0a.x; a.v[1] = (__bf16)c0a.y; a.v[2] = (__bf16)c0a.z; a.v[3] = (__bf16)c0a.w;
        a.v[4] = (__bf16)c0b.x; a.v[5] = (__bf16)c0b.y; a.v[6] = (__bf16)c0b.z; a.v[7] = (__bf16)c0b.w;
        a.v[8]  = (__bf16)c1a.x; a.v[9]  = (__bf16)c1a.y; a.v[10] = (__bf16)c1a.z; a.v[11] = (__bf16)c1a.w;
        a.v[12] = (__bf16)c1b.x; a.v[13] = (__bf16)c1b.y; a.v[14] = (__bf16)c1b.z; a.v[15] = (__bf16)c1b.w;
      } else {                                         // probs tail + zero pad
#pragma unroll
        for (int half = 0; half < 2; ++half) {
          const int base = k0 + half * 16;
#pragma unroll
          for (int j = 0; j < 8; ++j) {
            const int k = base + j;
            float v = 0.f;
            if (k < HD)            v = hrow[k];
            else if (k < HD + AAC) v = prow[k - HD];
            a.v[half * 8 + j] = (__bf16)v;
          }
        }
      }
      const __bf16* wb = reinterpret_cast<const __bf16*>(&wbuf_u[buf][0]);
      const int koff = hi ? 8 : 0;                     // B layout: two 8-K chunks
#pragma unroll
      for (int t = 0; t < 16; ++t) {
        const int n = t * 16 + col;
        const uint4v* bp = reinterpret_cast<const uint4v*>(wb + n * 32 + koff);
        Frag b; b.q[0] = bp[0]; b.q[1] = bp[2];
        acc[t] = wmma_bf16(a.v, b.v, acc[t]);
      }
      if (kb + 1 < KP1 / 32) wait_async0();
      __syncthreads();   // all waves done reading buf before it is re-staged
    }
  }
  // bias + silu -> LDS (C layout: VGPR v, lane L -> M = v + (L>=16?8:0), N = L&15)
#pragma unroll
  for (int t = 0; t < 16; ++t) {
    const int   n    = t * 16 + col;
    const float bias = b1[n];
#pragma unroll
    for (int vv = 0; vv < 8; ++vv) {
      float x = acc[t][vv] + bias;
      float s = x / (1.f + __expf(-x));
      const int r = wave * 16 + vv + (hi ? 8 : 0);
      zb[r * 256 + n] = (__bf16)s;
    }
  }
  __syncthreads();

  // ----------------- Phase 2: z2 = silu(z1 @ W2 + b2) -----------------
#pragma unroll
  for (int t = 0; t < 16; ++t) acc[t] = vzero;

  stage_slice(W2T, K2, 0, 0);
  wait_async0();
  __syncthreads();
  for (int kb = 0; kb < K2 / 32; ++kb) {
    const int buf = kb & 1;
    if (kb + 1 < K2 / 32) stage_slice(W2T, K2, kb + 1, 1 - buf);

    Frag a;
    const int r  = wave * 16 + col;
    const int k0 = kb * 32 + (hi ? 8 : 0);
    const uint4v* ap = reinterpret_cast<const uint4v*>(zb + r * 256 + k0);
    a.q[0] = ap[0];            // K chunk 0 (16B)
    a.q[1] = ap[2];            // K chunk at +16 elements
    const __bf16* wb = reinterpret_cast<const __bf16*>(&wbuf_u[buf][0]);
    const int koff = hi ? 8 : 0;
#pragma unroll
    for (int t = 0; t < 16; ++t) {
      const int n = t * 16 + col;
      const uint4v* bp = reinterpret_cast<const uint4v*>(wb + n * 32 + koff);
      Frag b; b.q[0] = bp[0]; b.q[1] = bp[2];
      acc[t] = wmma_bf16(a.v, b.v, acc[t]);
    }
    if (kb + 1 < K2 / 32) wait_async0();
    __syncthreads();
  }
  // write silu(z2) over the z1 LDS tile (reads finished at loop-exit barrier)
#pragma unroll
  for (int t = 0; t < 16; ++t) {
    const int   n    = t * 16 + col;
    const float bias = b2[n];
#pragma unroll
    for (int vv = 0; vv < 8; ++vv) {
      float x = acc[t][vv] + bias;
      float s = x / (1.f + __expf(-x));
      const int r = wave * 16 + vv + (hi ? 8 : 0);
      zb[r * 256 + n] = (__bf16)s;
    }
  }
  // stage entire head matrix (32x256 bf16 = 16KB) into wbuf[0]
  {
    const int n = threadIdx.x;                 // 256 threads x 64B
    const __bf16* src = WHT + (size_t)n * 32;
    const unsigned dst = wb_base[0] + (unsigned)(n * 32) * 2u;
#pragma unroll
    for (int j = 0; j < 4; ++j) async_load16(dst + j * 16u, src + j * 8);
  }
  wait_async0();
  __syncthreads();

  // ----------------- Phase 3: heads (tile0 = dchi, tile1 = sraw) -----------------
  const __bf16* whb = reinterpret_cast<const __bf16*>(&wbuf_u[0][0]);
  v8f acc3[2] = {vzero, vzero};
  for (int kb = 0; kb < K2 / 32; ++kb) {
    Frag a;
    const int r  = wave * 16 + col;
    const int k0 = kb * 32 + (hi ? 8 : 0);
    const uint4v* ap = reinterpret_cast<const uint4v*>(zb + r * 256 + k0);
    a.q[0] = ap[0];
    a.q[1] = ap[2];
    const int koffB = kb * 32 + (hi ? 8 : 0);
#pragma unroll
    for (int t = 0; t < 2; ++t) {
      const int n = t * 16 + col;
      const uint4v* bp = reinterpret_cast<const uint4v*>(whb + n * 256 + koffB);
      Frag b; b.q[0] = bp[0]; b.q[1] = bp[2];
      acc3[t] = wmma_bf16(a.v, b.v, acc3[t]);
    }
  }
#pragma unroll
  for (int vv = 0; vv < 8; ++vv) {
    const int gm = rowBase + vv + (hi ? 8 : 0);
    if (col < GMAXC)
      dchi[(size_t)gm * GMAXC + col] = tanhf(acc3[0][vv] + bt[col]) * 0.5f;
    if (col < NATOMC)
      sraw[(size_t)gm * NATOMC + col] = tanhf(acc3[1][vv] + bs[col]);
  }
}

// ---------------------------------------------------------------------------
// Forward kinematics: one thread per token, mirrors reference snapshot
// semantics (compute all rotated atoms per group, then scatter).
// ---------------------------------------------------------------------------
__global__ void fk_kernel(
    const float* __restrict__ X, const unsigned char* __restrict__ exists,
    const int* __restrict__ top_idx, const int* __restrict__ tors_axis,
    const int* __restrict__ aff_idx, const unsigned char* __restrict__ aff_mask,
    const int* __restrict__ G_counts, const unsigned char* __restrict__ scale_mask,
    const int* __restrict__ scale_parent, const float* __restrict__ dchi,
    const float* __restrict__ sraw, float* __restrict__ out, int M) {
  const int m = blockIdx.x * blockDim.x + threadIdx.x;
  if (m >= M) return;
  const int a = top_idx[m];

  float px[NATOMC], py[NATOMC], pz[NATOMC];
  const float* xp = X + (size_t)m * NATOMC * 3;
#pragma unroll
  for (int i = 0; i < NATOMC; ++i) { px[i] = xp[i*3]; py[i] = xp[i*3+1]; pz[i] = xp[i*3+2]; }
  bool ex[NATOMC];
#pragma unroll
  for (int i = 0; i < NATOMC; ++i) ex[i] = exists[(size_t)m * NATOMC + i] != 0;

  const int gc = G_counts[a];
#pragma unroll
  for (int gi = 0; gi < GMAXC; ++gi) {
    const int u = tors_axis[a * GMAXC * 2 + gi * 2 + 0];
    const int v = tors_axis[a * GMAXC * 2 + gi * 2 + 1];
    const bool ok = (u >= 0) && (v >= 0) && (gi < gc);
    const int iu = u > 0 ? u : 0;
    const int iv = v > 0 ? v : 0;
    const float pux = px[iu], puy = py[iu], puz = pz[iu];
    const float ax = px[iv] - pux, ay = py[iv] - puy, az = pz[iv] - puz;
    const float nrm = sqrtf(ax*ax + ay*ay + az*az);
    const float inv = 1.f / fmaxf(nrm, 1e-8f);
    const float anx = ax*inv, any = ay*inv, anz = az*inv;
    const float ang = ok ? dchi[(size_t)m * GMAXC + gi] : 0.f;
    const float cs = cosf(ang), sn = sinf(ang), omc = 1.f - cs;

    float nx[EMAXC], ny[EMAXC], nz[EMAXC];
    int   cv[EMAXC];
    bool  wm[EMAXC];
#pragma unroll
    for (int e = 0; e < EMAXC; ++e) {
      const int c  = aff_idx[a * GMAXC * EMAXC + gi * EMAXC + e];
      const int cc = c > 0 ? c : 0;
      const bool msk = aff_mask[a * GMAXC * EMAXC + gi * EMAXC + e] != 0;
      const float vx = px[cc] - pux, vy = py[cc] - puy, vz = pz[cc] - puz;
      const float crx = any*vz - anz*vy;
      const float cry = anz*vx - anx*vz;
      const float crz = anx*vy - any*vx;
      const float dt  = vx*anx + vy*any + vz*anz;
      nx[e] = pux + vx*cs + crx*sn + anx*dt*omc;
      ny[e] = puy + vy*cs + cry*sn + any*dt*omc;
      nz[e] = puz + vz*cs + crz*sn + anz*dt*omc;
      cv[e] = cc;
      wm[e] = msk && ex[cc];
    }
#pragma unroll
    for (int e = 0; e < EMAXC; ++e)
      if (wm[e]) { px[cv[e]] = nx[e]; py[cv[e]] = ny[e]; pz[cv[e]] = nz[e]; }
  }

  // scale step from snapshot
  float sx[NATOMC], sy[NATOMC], sz[NATOMC];
#pragma unroll
  for (int i = 0; i < NATOMC; ++i) { sx[i] = px[i]; sy[i] = py[i]; sz[i] = pz[i]; }
  float* op = out + (size_t)m * NATOMC * 3;
#pragma unroll
  for (int i = 0; i < NATOMC; ++i) {
    const int p  = scale_parent[a * NATOMC + i];
    const int pc = p > 0 ? p : 0;
    const float vx = sx[i] - sx[pc], vy = sy[i] - sy[pc], vz = sz[i] - sz[pc];
    const float r  = fmaxf(sqrtf(vx*vx + vy*vy + vz*vz), 1e-8f);
    const float se = fminf(fmaxf(sraw[(size_t)m * NATOMC + i], -1.f), 1.f) * 0.1f;
    const float sc = (r * (1.f + se)) / r;   // dir * r * (1 + s_eff)
    const float pax = sx[pc] + vx * sc;
    const float pay = sy[pc] + vy * sc;
    const float paz = sz[pc] + vz * sc;
    const bool W = (scale_mask[a * NATOMC + i] != 0) && ex[i] && ex[pc];
    op[i*3+0] = W ? pax : sx[i];
    op[i*3+1] = W ? pay : sy[i];
    op[i*3+2] = W ? paz : sz[i];
  }
}

// ---------------------------------------------------------------------------
extern "C" void kernel_launch(void* const* d_in, const int* in_sizes, int n_in,
                              void* d_out, int out_size, void* d_ws, size_t ws_size,
                              hipStream_t stream) {
  const float* h_   = (const float*)d_in[0];
  const float* ap   = (const float*)d_in[1];
  const float* X    = (const float*)d_in[2];
  const float* W1   = (const float*)d_in[3];
  const float* b1   = (const float*)d_in[4];
  const float* W2   = (const float*)d_in[5];
  const float* b2   = (const float*)d_in[6];
  const float* Wt   = (const float*)d_in[7];
  const float* bt   = (const float*)d_in[8];
  const float* Ws   = (const float*)d_in[9];
  const float* bs   = (const float*)d_in[10];
  const unsigned char* exists = (const unsigned char*)d_in[11];   // bool
  const int* top_idx = (const int*)d_in[12];
  const int* tors    = (const int*)d_in[13];
  const int* affi    = (const int*)d_in[14];
  const unsigned char* affm  = (const unsigned char*)d_in[15];    // bool
  const int* gcnt    = (const int*)d_in[16];
  const unsigned char* smask = (const unsigned char*)d_in[17];    // bool
  const int* sparent = (const int*)d_in[18];

  const int M = in_sizes[0] / HD;   // B*N tokens (65536)

  // workspace layout (256B-aligned offsets)
  char* ws = (char*)d_ws;
  unsigned short* w1t = (unsigned short*)(ws + 0);            // 147456 B
  unsigned short* w2t = (unsigned short*)(ws + 147456);       // 131072 B
  unsigned short* wht = (unsigned short*)(ws + 278528);       //  16384 B
  float* dchi = (float*)(ws + 294912);                        // M*4*4  B
  float* sraw = (float*)(ws + 294912 + (size_t)M * GMAXC * sizeof(float));

  pack_weights_kernel<<<(HD * KP1 + HD * K2 + 32 * K2 + 255) / 256, 256, 0, stream>>>(
      W1, W2, Wt, Ws, w1t, w2t, wht);
  mlp_heads_kernel<<<M / 128, 256, 0, stream>>>(
      h_, ap, b1, b2, bt, bs, w1t, w2t, wht, dchi, sraw);
  fk_kernel<<<(M + 255) / 256, 256, 0, stream>>>(
      X, exists, top_idx, tors, affi, affm, gcnt, smask, sparent, dchi, sraw,
      (float*)d_out, M);
}
